// TransformerEncoder_33054068310657
// MI455X (gfx1250) — matrix operand order repaired and checked by reading, not compiled, not run on hardware
//
#include <hip/hip_runtime.h>
#include <stdint.h>

typedef __bf16       v16bf __attribute__((ext_vector_type(16)));
typedef float        v8f   __attribute__((ext_vector_type(8)));
typedef unsigned int u32x4 __attribute__((ext_vector_type(4)));

#define B_   2
#define S_   2048
#define D_   1024
#define H_   16
#define HD_  64
#define L_   6
#define DFF_ 4096
#define BS_  (B_*S_)

__device__ __forceinline__ unsigned short f2bf(float f){
  union{float f; unsigned u;} x; x.f=f;
  unsigned u = x.u;
  unsigned r = u + 0x7FFFu + ((u>>16)&1u);   // round-to-nearest-even
  return (unsigned short)(r>>16);
}

union FragBF  { v16bf v; unsigned short u[16]; u32x4 q[2]; };
union Pack16  { u32x4 q[2]; unsigned short u[16]; };

__device__ __forceinline__ u32x4 ld16(const unsigned short* p){
  return *reinterpret_cast<const u32x4*>(p);
}
__device__ __forceinline__ void st16(unsigned short* p, u32x4 v){
  *reinterpret_cast<u32x4*>(p) = v;
}

// 32 bytes: global -> LDS via the CDNA5 async DMA path (ASYNCcnt-tracked,
// bypasses VGPRs).  INST_OFFSET is added to BOTH the global and LDS address,
// so chunk 1 is the offset:16 form of the same operands.
__device__ __forceinline__ void async_cp32(unsigned lds_addr,
                                           const unsigned short* gptr){
  asm volatile(
      "global_load_async_to_lds_b128 %0, %1, off\n\t"
      "global_load_async_to_lds_b128 %0, %1, off offset:16"
      :: "v"(lds_addr), "v"(gptr) : "memory");
}
__device__ __forceinline__ void async_wait0(){
  asm volatile("s_wait_asynccnt 0" ::: "memory");
}
__device__ __forceinline__ unsigned lds_off(const void* p){
  return (unsigned)(uintptr_t)p;   // LDS flat addresses are addr[31:0]
}

// ---------------------------------------------------------------------------
// Embedding + positional encoding -> f32 residual stream + bf16 activations
// ---------------------------------------------------------------------------
__global__ __launch_bounds__(256) void k_embed(
    const int* __restrict__ x, const float* __restrict__ emb,
    const float* __restrict__ pe, float* __restrict__ hf,
    unsigned short* __restrict__ hb)
{
  int row = blockIdx.x;              // 0..BS-1
  int s   = row % S_;
  int tok = x[row];
  int tid = threadIdx.x;
  #pragma unroll
  for(int i=0;i<4;i++){
    int c = tid + i*256;
    float v = emb[(size_t)tok*D_ + c]*32.0f + pe[(size_t)s*D_ + c]; // sqrt(1024)=32
    hf[(size_t)row*D_ + c] = v;
    hb[(size_t)row*D_ + c] = f2bf(v);
  }
}

// ---------------------------------------------------------------------------
// Weight repack: wq/wk/wv [H][D][HD] f32 -> packed [D][3*D] bf16
// col n = proj*1024 + h*64 + k
// ---------------------------------------------------------------------------
__global__ __launch_bounds__(256) void k_conv_qkv(
    const float* __restrict__ wq, const float* __restrict__ wk,
    const float* __restrict__ wv, unsigned short* __restrict__ out)
{
  size_t i = (size_t)blockIdx.x*256 + threadIdx.x;
  if(i >= (size_t)3*D_*D_) return;
  int d    = (int)(i / (3*D_));
  int n    = (int)(i % (3*D_));
  int proj = n >> 10;
  int rest = n & 1023;
  int hh   = rest >> 6;
  int kk   = rest & 63;
  const float* w = (proj==0)? wq : ((proj==1)? wk : wv);
  out[i] = f2bf(w[((size_t)hh*D_ + d)*HD_ + kk]);
}

__global__ __launch_bounds__(256) void k_conv(
    const float* __restrict__ src, unsigned short* __restrict__ dst, size_t n)
{
  size_t i = (size_t)blockIdx.x*256 + threadIdx.x;
  if(i<n) dst[i]=f2bf(src[i]);
}

// ---------------------------------------------------------------------------
// bf16 GEMM: C[M][N] = A[M][K]*B[K][N] (+bias)(+res)(relu).  Macro tile
// 128x128, K-step 32, 8 waves as 4(M)x2(N); wave tile 32x64 = 2x4 WMMA accs.
// A tile: double-buffered LDS filled by global_load_async_to_lds_b128 --
// the next slab's DMA overlaps the whole WMMA block.  B tile: register-
// staged with an LDS transpose ([n][k]) so fragment loads are ds_load_b128.
// ---------------------------------------------------------------------------
__global__ __launch_bounds__(256) void k_gemm_bf16(
    const unsigned short* __restrict__ A,   // [M][K] bf16
    const unsigned short* __restrict__ Bm,  // [K][N] bf16
    int N, int K,
    const float* __restrict__ bias,         // [N] or null
    const float* __restrict__ res,          // [M][N] or null
    int relu,
    float* __restrict__ outf,               // [M][N] or null
    unsigned short* __restrict__ outb)      // [M][N] or null
{
  __shared__ __align__(16) unsigned short As[2][128][32]; // [buf][m][k]
  __shared__ __align__(16) unsigned short Bs[128][32];    // [n][k] (transposed!)
  const int tid  = threadIdx.x;
  const int m0   = blockIdx.y*128;
  const int n0   = blockIdx.x*128;
  const int w    = tid>>5;
  const int lane = tid&31;
  const int lo   = lane&15, half = lane>>4;
  const int wm   = (w&3)*32, wn = (w>>2)*64;

  v8f acc[2][4];
  #pragma unroll
  for(int i=0;i<2;i++)
    #pragma unroll
    for(int j=0;j<4;j++)
      #pragma unroll
      for(int e=0;e<8;e++) acc[i][j][e]=0.f;

  // staging assignments
  const int arow = tid>>1, ak0 = (tid&1)*16;   // A: 32B contiguous per thread
  const int bk   = tid>>3, bn0 = (tid&7)*16;   // B: 16 contiguous n per thread

  // --- prologue: async A slab 0, register B slab 0 ---
  async_cp32(lds_off(&As[0][arow][ak0]), A + (size_t)(m0+arow)*K + ak0);
  Pack16 bR;
  {
    const unsigned short* bp = Bm + (size_t)bk*N + n0 + bn0;
    bR.q[0] = ld16(bp); bR.q[1] = ld16(bp+8);
  }
  #pragma unroll
  for(int i=0;i<16;i++) Bs[bn0+i][bk] = bR.u[i];   // transpose on store
  async_wait0();
  __syncthreads();

  int p = 0;
  for(int kb=0; kb<K; kb+=32){
    const bool last = (kb+32>=K);
    Pack16 bN;
    if(!last){
      // A: async DMA straight into the other LDS buffer (overlaps compute)
      async_cp32(lds_off(&As[p^1][arow][ak0]),
                 A + (size_t)(m0+arow)*K + (kb+32) + ak0);
      // B: prefetch to registers (overlaps compute)
      const unsigned short* bp = Bm + (size_t)(kb+32+bk)*N + n0 + bn0;
      bN.q[0] = ld16(bp); bN.q[1] = ld16(bp+8);
      if(kb+64<K)
        __builtin_prefetch(Bm + (size_t)(kb+64+bk)*N + n0 + bn0, 0, 3);
    }

    // fragment loads: all 16-byte LDS vector reads
    FragBF a[2], bfr[4];
    #pragma unroll
    for(int mi=0;mi<2;mi++){
      int m = wm + 16*mi + lo;
      a[mi].q[0] = ld16(&As[p][m][half*8]);        // K 0..7  / 8..15
      a[mi].q[1] = ld16(&As[p][m][half*8+16]);     // K 16..23 / 24..31
    }
    #pragma unroll
    for(int ni=0;ni<4;ni++){
      int n = wn + 16*ni + lo;
      bfr[ni].q[0] = ld16(&Bs[n][half*8]);         // K 0..7  / 8..15
      bfr[ni].q[1] = ld16(&Bs[n][half*8+16]);      // K 16..23 / 24..31
    }
    #pragma unroll
    for(int mi=0;mi<2;mi++)
      #pragma unroll
      for(int ni=0;ni<4;ni++)
        acc[mi][ni] = __builtin_amdgcn_wmma_f32_16x16x32_bf16(
            false, a[mi].v, false, bfr[ni].v, (short)0, acc[mi][ni], false, false);

    if(!last){
      __syncthreads();                  // everyone done reading Bs / As[p]
      #pragma unroll
      for(int i=0;i<16;i++) Bs[bn0+i][bk] = bN.u[i];
      async_wait0();                    // As[p^1] DMA landed (per wave)
      __syncthreads();
      p ^= 1;
    }
  }

  // epilogue: bias / residual / relu, f32 and/or bf16 stores
  #pragma unroll
  for(int mi=0;mi<2;mi++)
    #pragma unroll
    for(int ni=0;ni<4;ni++){
      int col = n0 + wn + 16*ni + lo;
      float bv = bias ? bias[col] : 0.f;
      #pragma unroll
      for(int r=0;r<8;r++){
        int row = m0 + wm + 16*mi + half*8 + r;
        float v = acc[mi][ni][r] + bv;
        if(res)  v += res[(size_t)row*N + col];
        if(relu) v = v>0.f ? v : 0.f;
        if(outf) outf[(size_t)row*N + col] = v;
        if(outb) outb[(size_t)row*N + col] = f2bf(v);
      }
    }
}

// ---------------------------------------------------------------------------
// Flash attention: one (b,h) x 64 query rows per block; 4 waves x 16 rows.
// qkv packed bf16 [BS][3072]: q @ h*64, k @ 1024+h*64, v @ 2048+h*64
// K tile staged via async DMA; V tile register-transposed (overlaps it).
// ---------------------------------------------------------------------------
__global__ __launch_bounds__(128) void k_flash(
    const unsigned short* __restrict__ qkv,
    const int* __restrict__ mask,            // [B][S]
    unsigned short* __restrict__ attn)       // [BS][D] bf16
{
  __shared__ __align__(16) unsigned short Ks[32][64];    // [key][hd]
  __shared__ __align__(16) unsigned short Vs[64][32];    // [hd][key] (transposed)
  __shared__ __align__(16) unsigned short Pb[4][16][32]; // per-wave P transpose
  const int bh   = blockIdx.x;
  const int b    = bh / H_, h = bh % H_;
  const int tid  = threadIdx.x;
  const int w    = tid>>5, lane = tid&31;
  const int lo   = lane&15, half = lane>>4;
  const int qbase= blockIdx.y*64 + w*16;

  // Q fragments (16 rows x 64 hd -> two 16x32 A-frags), vector global loads
  FragBF qf[2];
  {
    size_t base = (size_t)(b*S_ + qbase + lo)*3072 + h*HD_;
    #pragma unroll
    for(int f=0; f<2; f++){
      qf[f].q[0] = ld16(&qkv[base + f*32 + half*8]);
      qf[f].q[1] = ld16(&qkv[base + f*32 + half*8 + 16]);
    }
  }

  float m_i[8], l_i[8];
  v8f accO[4];
  #pragma unroll
  for(int r=0;r<8;r++){ m_i[r] = -3.0e38f; l_i[r]=0.f; }
  #pragma unroll
  for(int j=0;j<4;j++)
    #pragma unroll
    for(int e=0;e<8;e++) accO[j][e]=0.f;

  const int si = tid>>2;        // 0..31 key idx for staging
  const int sj = (tid&3)*16;    // hd start for staging

  for(int kt=0; kt<S_; kt+=32){
    { // K tile: async global->LDS DMA; V tile: register transpose (overlaps)
      size_t kbase = (size_t)(b*S_ + kt + si)*3072 + 1024 + h*HD_ + sj;
      size_t vbase = (size_t)(b*S_ + kt + si)*3072 + 2048 + h*HD_ + sj;
      async_cp32(lds_off(&Ks[si][sj]), &qkv[kbase]);
      Pack16 vv; vv.q[0] = ld16(&qkv[vbase]); vv.q[1] = ld16(&qkv[vbase+8]);
      #pragma unroll
      for(int i=0;i<16;i++) Vs[sj+i][si] = vv.u[i];
      async_wait0();
    }
    __syncthreads();

    // scores: two 16-col subtiles, K-dim = hd = 64 (2 chained WMMAs each)
    v8f s[2];
    #pragma unroll
    for(int n=0;n<2;n++){
      FragBF k0, k1;
      int key = n*16 + lo;
      k0.q[0] = ld16(&Ks[key][half*8]);
      k0.q[1] = ld16(&Ks[key][half*8+16]);
      k1.q[0] = ld16(&Ks[key][32+half*8]);
      k1.q[1] = ld16(&Ks[key][32+half*8+16]);
      v8f z;
      #pragma unroll
      for(int e=0;e<8;e++) z[e]=0.f;
      z = __builtin_amdgcn_wmma_f32_16x16x32_bf16(false, qf[0].v, false, k0.v, (short)0, z, false, false);
      z = __builtin_amdgcn_wmma_f32_16x16x32_bf16(false, qf[1].v, false, k1.v, (short)0, z, false, false);
      s[n] = z;
    }
    const float sc = 0.125f;                  // 1/sqrt(64)
    int mv0 = mask[b*S_ + kt + lo];
    int mv1 = mask[b*S_ + kt + 16 + lo];
    #pragma unroll
    for(int e=0;e<8;e++){
      s[0][e] = mv0 ? s[0][e]*sc : -3.0e38f;
      s[1][e] = mv1 ? s[1][e]*sc : -3.0e38f;
    }

    // online softmax; rows live in 16-lane halves of the wave
    float alpha[8];
    #pragma unroll
    for(int r=0;r<8;r++){
      float t = fmaxf(s[0][r], s[1][r]);
      for(int msk=1; msk<16; msk<<=1) t = fmaxf(t, __shfl_xor(t, msk, 32));
      float mn = fmaxf(m_i[r], t);
      alpha[r] = __expf(m_i[r] - mn);
      m_i[r] = mn;
      float p0 = __expf(s[0][r]-mn);
      float p1 = __expf(s[1][r]-mn);
      s[0][r]=p0; s[1][r]=p1;
      float rs = p0+p1;
      for(int msk=1; msk<16; msk<<=1) rs += __shfl_xor(rs, msk, 32);
      l_i[r] = l_i[r]*alpha[r] + rs;
    }

    // transpose P through LDS into A-frag layout
    #pragma unroll
    for(int r=0;r<8;r++){
      Pb[w][half*8+r][lo]      = f2bf(s[0][r]);
      Pb[w][half*8+r][16+lo]   = f2bf(s[1][r]);
    }
    __syncthreads();

    FragBF pf;
    pf.q[0] = ld16(&Pb[w][lo][half*8]);
    pf.q[1] = ld16(&Pb[w][lo][half*8+16]);

    // O := O*alpha + P(16x32) x V(32x64)
    #pragma unroll
    for(int j=0;j<4;j++){
      FragBF vf;
      vf.q[0] = ld16(&Vs[16*j+lo][half*8]);
      vf.q[1] = ld16(&Vs[16*j+lo][half*8+16]);
      #pragma unroll
      for(int r=0;r<8;r++) accO[j][r] *= alpha[r];
      accO[j] = __builtin_amdgcn_wmma_f32_16x16x32_bf16(
          false, pf.v, false, vf.v, (short)0, accO[j], false, false);
    }
    __syncthreads();
  }

  #pragma unroll
  for(int j=0;j<4;j++)
    #pragma unroll
    for(int r=0;r<8;r++){
      int q = qbase + half*8 + r;
      float v = accO[j][r] / l_i[r];
      attn[(size_t)(b*S_ + q)*D_ + h*HD_ + 16*j + lo] = f2bf(v);
    }
}

// ---------------------------------------------------------------------------
// LayerNorm (row per block) -> f32 residual stream + bf16 activations
// ---------------------------------------------------------------------------
__global__ __launch_bounds__(256) void k_ln(
    const float* __restrict__ x, const float* __restrict__ sc,
    const float* __restrict__ bi, float* __restrict__ outf,
    unsigned short* __restrict__ outb)
{
  __shared__ float red[256];
  int row = blockIdx.x; int tid = threadIdx.x;
  const float* xr = x + (size_t)row*D_;
  float s=0.f, s2=0.f, v[4];
  #pragma unroll
  for(int i=0;i<4;i++){ v[i]=xr[tid + i*256]; s+=v[i]; s2+=v[i]*v[i]; }
  red[tid]=s; __syncthreads();
  for(int o=128;o>0;o>>=1){ if(tid<o) red[tid]+=red[tid+o]; __syncthreads(); }
  float mean = red[0]/(float)D_; __syncthreads();
  red[tid]=s2; __syncthreads();
  for(int o=128;o>0;o>>=1){ if(tid<o) red[tid]+=red[tid+o]; __syncthreads(); }
  float var = red[0]/(float)D_ - mean*mean;
  float inv = rsqrtf(var + 1e-5f);
  #pragma unroll
  for(int i=0;i<4;i++){
    int c = tid + i*256;
    float y = (v[i]-mean)*inv*sc[c] + bi[c];
    outf[(size_t)row*D_ + c] = y;
    outb[(size_t)row*D_ + c] = f2bf(y);
  }
}

// ---------------------------------------------------------------------------
extern "C" void kernel_launch(void* const* d_in, const int* in_sizes, int n_in,
                              void* d_out, int out_size, void* d_ws, size_t ws_size,
                              hipStream_t stream)
{
  (void)in_sizes; (void)n_in; (void)out_size; (void)ws_size;
  const int*   x    = (const int*)  d_in[0];
  const int*   mask = (const int*)  d_in[1];
  const float* emb  = (const float*)d_in[2];
  const float* pe   = (const float*)d_in[3];
  const float* wq   = (const float*)d_in[4];
  const float* wk   = (const float*)d_in[5];
  const float* wv   = (const float*)d_in[6];
  const float* wo   = (const float*)d_in[7];
  const float* bo   = (const float*)d_in[8];
  const float* ln1s = (const float*)d_in[9];
  const float* ln1b = (const float*)d_in[10];
  const float* w1   = (const float*)d_in[11];
  const float* b1   = (const float*)d_in[12];
  const float* w2   = (const float*)d_in[13];
  const float* b2   = (const float*)d_in[14];
  const float* ln2s = (const float*)d_in[15];
  const float* ln2b = (const float*)d_in[16];

  char* ws = (char*)d_ws;
  size_t off = 0;
  auto alloc = [&](size_t bytes)->char*{
    char* p = ws + off;
    off += (bytes + 255) & ~(size_t)255;
    return p;
  };
  float*          hF   = (float*)         alloc((size_t)BS_*D_*4);
  float*          tmpF = (float*)         alloc((size_t)BS_*D_*4);
  unsigned short* hB   = (unsigned short*)alloc((size_t)BS_*D_*2);
  unsigned short* qkvB = (unsigned short*)alloc((size_t)BS_*3*D_*2);
  unsigned short* atB  = (unsigned short*)alloc((size_t)BS_*D_*2);
  unsigned short* ffB  = (unsigned short*)alloc((size_t)BS_*DFF_*2);
  unsigned short* Wqkv = (unsigned short*)alloc((size_t)3*D_*D_*2);
  unsigned short* Wo   = (unsigned short*)alloc((size_t)D_*D_*2);
  unsigned short* W1   = (unsigned short*)alloc((size_t)D_*DFF_*2);
  unsigned short* W2   = (unsigned short*)alloc((size_t)DFF_*D_*2);

  k_embed<<<BS_, 256, 0, stream>>>(x, emb, pe, hF, hB);

  for(int l=0; l<L_; l++){
    // --- weight repack (fp32 -> bf16) for this layer ---
    k_conv_qkv<<<(3*D_*D_+255)/256, 256, 0, stream>>>(
        wq + (size_t)l*H_*D_*HD_, wk + (size_t)l*H_*D_*HD_,
        wv + (size_t)l*H_*D_*HD_, Wqkv);
    k_conv<<<((size_t)D_*D_+255)/256, 256, 0, stream>>>(wo + (size_t)l*D_*D_,   Wo, (size_t)D_*D_);
    k_conv<<<((size_t)D_*DFF_+255)/256, 256, 0, stream>>>(w1 + (size_t)l*D_*DFF_, W1, (size_t)D_*DFF_);
    k_conv<<<((size_t)DFF_*D_+255)/256, 256, 0, stream>>>(w2 + (size_t)l*DFF_*D_, W2, (size_t)DFF_*D_);

    // --- QKV projection: [BS,1024] x [1024,3072] -> qkv bf16 ---
    k_gemm_bf16<<<dim3(3*D_/128, BS_/128), 256, 0, stream>>>(
        hB, Wqkv, 3*D_, D_, nullptr, nullptr, 0, nullptr, qkvB);

    // --- flash attention -> atB bf16 [BS,1024] ---
    k_flash<<<dim3(B_*H_, S_/64), 128, 0, stream>>>(qkvB, mask, atB);

    // --- output projection + bias + residual -> tmpF ---
    k_gemm_bf16<<<dim3(D_/128, BS_/128), 256, 0, stream>>>(
        atB, Wo, D_, D_, bo + (size_t)l*D_, hF, 0, tmpF, nullptr);

    // --- LN1 -> hF (residual) + hB (bf16) ---
    k_ln<<<BS_, 256, 0, stream>>>(tmpF, ln1s + (size_t)l*D_, ln1b + (size_t)l*D_, hF, hB);

    // --- FFN up + bias + relu -> ffB bf16 ---
    k_gemm_bf16<<<dim3(DFF_/128, BS_/128), 256, 0, stream>>>(
        hB, W1, DFF_, D_, b1 + (size_t)l*DFF_, nullptr, 1, nullptr, ffB);

    // --- FFN down + bias + residual -> tmpF ---
    k_gemm_bf16<<<dim3(D_/128, BS_/128), 256, 0, stream>>>(
        ffB, W2, D_, DFF_, b2 + (size_t)l*D_, hF, 0, tmpF, nullptr);

    // --- LN2 -> hF + hB ---
    k_ln<<<BS_, 256, 0, stream>>>(tmpF, ln2s + (size_t)l*D_, ln2b + (size_t)l*D_, hF, hB);
  }

  hipMemcpyAsync(d_out, hF, (size_t)BS_*D_*4, hipMemcpyDeviceToDevice, stream);
}
